// SmallMLP_49950469652693
// MI455X (gfx1250) — 16-bit and 8-bit matrix operand order repaired and checked by reading, compiled, not run on hardware
//
#include <hip/hip_runtime.h>
#include <hip/hip_bf16.h>

typedef __attribute__((ext_vector_type(16))) _Float16 v16h;
typedef __attribute__((ext_vector_type(8)))  _Float16 v8h;
typedef __attribute__((ext_vector_type(8)))  float    v8f;
typedef __attribute__((ext_vector_type(8)))  int      v8i;
typedef __attribute__((ext_vector_type(2)))  unsigned long long v2u64;

#define B_SZ  1024
#define D_IN  4096
#define D_H   8192
#define D_OUT 4096

__device__ __forceinline__ float ternf(float w) {
    return w > 0.001f ? 1.0f : (w < -0.001f ? -1.0f : 0.0f);
}


__global__ __launch_bounds__(256) void k_tern_f16(const float* __restrict__ w,
                                                  _Float16* __restrict__ o) {
    int i = (blockIdx.x * 256 + threadIdx.x) * 8;
    float4 a = *(const float4*)(w + i);
    float4 b = *(const float4*)(w + i + 4);
    v8h r;
    r[0] = (_Float16)ternf(a.x); r[1] = (_Float16)ternf(a.y);
    r[2] = (_Float16)ternf(a.z); r[3] = (_Float16)ternf(a.w);
    r[4] = (_Float16)ternf(b.x); r[5] = (_Float16)ternf(b.y);
    r[6] = (_Float16)ternf(b.z); r[7] = (_Float16)ternf(b.w);
    *(v8h*)(o + i) = r;
}

__global__ __launch_bounds__(256) void k_cvt_f16(const float* __restrict__ x,
                                                 _Float16* __restrict__ o) {
    int i = (blockIdx.x * 256 + threadIdx.x) * 8;
    float4 a = *(const float4*)(x + i);
    float4 b = *(const float4*)(x + i + 4);
    v8h r;
    r[0] = (_Float16)a.x; r[1] = (_Float16)a.y; r[2] = (_Float16)a.z; r[3] = (_Float16)a.w;
    r[4] = (_Float16)b.x; r[5] = (_Float16)b.y; r[6] = (_Float16)b.z; r[7] = (_Float16)b.w;
    *(v8h*)(o + i) = r;
}

__global__ __launch_bounds__(256) void k_tern_i8(const float* __restrict__ w,
                                                 signed char* __restrict__ o) {
    int i = (blockIdx.x * 256 + threadIdx.x) * 8;
    float4 a = *(const float4*)(w + i);
    float4 b = *(const float4*)(w + i + 4);
    union { unsigned long long u; signed char c[8]; } p;
    p.c[0] = (signed char)ternf(a.x); p.c[1] = (signed char)ternf(a.y);
    p.c[2] = (signed char)ternf(a.z); p.c[3] = (signed char)ternf(a.w);
    p.c[4] = (signed char)ternf(b.x); p.c[5] = (signed char)ternf(b.y);
    p.c[6] = (signed char)ternf(b.z); p.c[7] = (signed char)ternf(b.w);
    *(unsigned long long*)(o + i) = p.u;
}

__global__ __launch_bounds__(256) void k_gemm1(const _Float16* __restrict__ A,
                                               const _Float16* __restrict__ Wt,
                                               const float* __restrict__ b1,
                                               const float* __restrict__ rmaxp,
                                               signed char* __restrict__ H)
{
    const int lane  = threadIdx.x & 31;
    const int wave  = threadIdx.x >> 5;
    const int lh    = lane & 15;
    const int hi    = lane >> 4;
    const int waveM = wave >> 2;
    const int waveN = wave & 3;
    const int m0 = blockIdx.y * 64  + waveM * 32;
    const int n0 = blockIdx.x * 256 + waveN * 64;

    v8f acc[2][4];
#pragma unroll
    for (int i = 0; i < 2; ++i)
#pragma unroll
        for (int j = 0; j < 4; ++j)
            acc[i][j] = (v8f){0.f,0.f,0.f,0.f,0.f,0.f,0.f,0.f};

    const _Float16* arow0 = A + (size_t)(m0 + lh) * D_IN + hi * 16;
    const _Float16* arow1 = arow0 + (size_t)16 * D_IN;
    const _Float16* brow0 = Wt + (size_t)(n0 + 0 * 16 + lh) * D_IN + hi * 16;
    const _Float16* brow1 = Wt + (size_t)(n0 + 1 * 16 + lh) * D_IN + hi * 16;
    const _Float16* brow2 = Wt + (size_t)(n0 + 2 * 16 + lh) * D_IN + hi * 16;
    const _Float16* brow3 = Wt + (size_t)(n0 + 3 * 16 + lh) * D_IN + hi * 16;

    for (int k0 = 0; k0 < D_IN; k0 += 256) {
        __builtin_prefetch(arow0 + k0 + 256, 0, 3);
        __builtin_prefetch(arow1 + k0 + 256, 0, 3);
        __builtin_prefetch(brow0 + k0 + 256, 0, 3);
        __builtin_prefetch(brow1 + k0 + 256, 0, 3);
        __builtin_prefetch(brow2 + k0 + 256, 0, 3);
        __builtin_prefetch(brow3 + k0 + 256, 0, 3);
#pragma unroll
        for (int kk = 0; kk < 256; kk += 32) {
            const int k = k0 + kk;
            union { v16h v; v8h h[2]; } a0, a1;
            a0.h[0] = *(const v8h*)(arow0 + k);
            a0.h[1] = *(const v8h*)(arow0 + k + 8);
            a1.h[0] = *(const v8h*)(arow1 + k);
            a1.h[1] = *(const v8h*)(arow1 + k + 8);

            v16h bj;
            bj = *(const v16h*)(brow0 + k);
            acc[0][0] = __builtin_amdgcn_wmma_f32_16x16x32_f16(false, a0.v, false, bj, (short)0, acc[0][0], false, false);
            acc[1][0] = __builtin_amdgcn_wmma_f32_16x16x32_f16(false, a1.v, false, bj, (short)0, acc[1][0], false, false);
            bj = *(const v16h*)(brow1 + k);
            acc[0][1] = __builtin_amdgcn_wmma_f32_16x16x32_f16(false, a0.v, false, bj, (short)0, acc[0][1], false, false);
            acc[1][1] = __builtin_amdgcn_wmma_f32_16x16x32_f16(false, a1.v, false, bj, (short)0, acc[1][1], false, false);
            bj = *(const v16h*)(brow2 + k);
            acc[0][2] = __builtin_amdgcn_wmma_f32_16x16x32_f16(false, a0.v, false, bj, (short)0, acc[0][2], false, false);
            acc[1][2] = __builtin_amdgcn_wmma_f32_16x16x32_f16(false, a1.v, false, bj, (short)0, acc[1][2], false, false);
            bj = *(const v16h*)(brow3 + k);
            acc[0][3] = __builtin_amdgcn_wmma_f32_16x16x32_f16(false, a0.v, false, bj, (short)0, acc[0][3], false, false);
            acc[1][3] = __builtin_amdgcn_wmma_f32_16x16x32_f16(false, a1.v, false, bj, (short)0, acc[1][3], false, false);
        }
    }

    const float scale = rmaxp[0] * (1.0f / 127.0f);
    const float thr   = 1.5f * scale;

#pragma unroll
    for (int i = 0; i < 2; ++i) {
        const int mt = m0 + i * 16;
#pragma unroll
        for (int j = 0; j < 4; ++j) {
            const int   col  = n0 + j * 16 + lh;
            const float bias = b1[col];
#pragma unroll
            for (int r = 0; r < 8; ++r) {
                const int   row = mt + hi * 8 + r;
                const float h   = acc[i][j][r] + bias;
                const signed char t = (h >= thr) ? 1 : ((h <= -thr) ? -1 : 0);
                H[(size_t)row * D_H + col] = t;
            }
        }
    }
}

__global__ __launch_bounds__(256) void k_gemm2(const signed char* __restrict__ A,
                                               const signed char* __restrict__ Wt,
                                               const float* __restrict__ b2,
                                               const float* __restrict__ rmaxp,
                                               float* __restrict__ Out)
{
    const int lane  = threadIdx.x & 31;
    const int wave  = threadIdx.x >> 5;
    const int lh    = lane & 15;
    const int hi    = lane >> 4;
    const int waveM = wave >> 2;
    const int waveN = wave & 3;
    const int m0 = blockIdx.y * 64  + waveM * 32;
    const int n0 = blockIdx.x * 256 + waveN * 64;

    v8i acc[2][4];
#pragma unroll
    for (int i = 0; i < 2; ++i)
#pragma unroll
        for (int j = 0; j < 4; ++j)
            acc[i][j] = (v8i){0,0,0,0,0,0,0,0};

    const signed char* arow0 = A + (size_t)(m0 + lh) * D_H + hi * 16;
    const signed char* arow1 = arow0 + (size_t)16 * D_H;
    const signed char* brow0 = Wt + (size_t)(n0 + 0 * 16 + lh) * D_H + hi * 16;
    const signed char* brow1 = Wt + (size_t)(n0 + 1 * 16 + lh) * D_H + hi * 16;
    const signed char* brow2 = Wt + (size_t)(n0 + 2 * 16 + lh) * D_H + hi * 16;
    const signed char* brow3 = Wt + (size_t)(n0 + 3 * 16 + lh) * D_H + hi * 16;

    for (int k0 = 0; k0 < D_H; k0 += 512) {
        __builtin_prefetch(arow0 + k0 + 512, 0, 3);
        __builtin_prefetch(arow1 + k0 + 512, 0, 3);
        __builtin_prefetch(brow0 + k0 + 512, 0, 3);
        __builtin_prefetch(brow1 + k0 + 512, 0, 3);
        __builtin_prefetch(brow2 + k0 + 512, 0, 3);
        __builtin_prefetch(brow3 + k0 + 512, 0, 3);
#pragma unroll
        for (int kk = 0; kk < 512; kk += 64) {
            const int k = k0 + kk;
            union { v8i v; unsigned long long q[4]; } a0, a1;
            a0.q[0] = *(const unsigned long long*)(arow0 + k);
            a0.q[1] = *(const unsigned long long*)(arow0 + k + 8);
            a0.q[2] = *(const unsigned long long*)(arow0 + k + 32);
            a0.q[3] = *(const unsigned long long*)(arow0 + k + 40);
            a1.q[0] = *(const unsigned long long*)(arow1 + k);
            a1.q[1] = *(const unsigned long long*)(arow1 + k + 8);
            a1.q[2] = *(const unsigned long long*)(arow1 + k + 32);
            a1.q[3] = *(const unsigned long long*)(arow1 + k + 40);

            union { v8i v; v2u64 d[2]; } b;
            b.d[0] = *(const v2u64*)(brow0 + k);
            b.d[1] = *(const v2u64*)(brow0 + k + 32);
            acc[0][0] = __builtin_amdgcn_wmma_i32_16x16x64_iu8(true, a0.v, true, b.v, acc[0][0], false, false);
            acc[1][0] = __builtin_amdgcn_wmma_i32_16x16x64_iu8(true, a1.v, true, b.v, acc[1][0], false, false);
            b.d[0] = *(const v2u64*)(brow1 + k);
            b.d[1] = *(const v2u64*)(brow1 + k + 32);
            acc[0][1] = __builtin_amdgcn_wmma_i32_16x16x64_iu8(true, a0.v, true, b.v, acc[0][1], false, false);
            acc[1][1] = __builtin_amdgcn_wmma_i32_16x16x64_iu8(true, a1.v, true, b.v, acc[1][1], false, false);
            b.d[0] = *(const v2u64*)(brow2 + k);
            b.d[1] = *(const v2u64*)(brow2 + k + 32);
            acc[0][2] = __builtin_amdgcn_wmma_i32_16x16x64_iu8(true, a0.v, true, b.v, acc[0][2], false, false);
            acc[1][2] = __builtin_amdgcn_wmma_i32_16x16x64_iu8(true, a1.v, true, b.v, acc[1][2], false, false);
            b.d[0] = *(const v2u64*)(brow3 + k);
            b.d[1] = *(const v2u64*)(brow3 + k + 32);
            acc[0][3] = __builtin_amdgcn_wmma_i32_16x16x64_iu8(true, a0.v, true, b.v, acc[0][3], false, false);
            acc[1][3] = __builtin_amdgcn_wmma_i32_16x16x64_iu8(true, a1.v, true, b.v, acc[1][3], false, false);
        }
    }

    const float scale = rmaxp[0] * (1.0f / 127.0f);
#pragma unroll
    for (int i = 0; i < 2; ++i) {
        const int mt = m0 + i * 16;
#pragma unroll
        for (int j = 0; j < 4; ++j) {
            const int   col  = n0 + j * 16 + lh;
            const float bias = b2[col];
#pragma unroll
            for (int r = 0; r < 8; ++r) {
                const int row = mt + hi * 8 + r;
                Out[(size_t)row * D_OUT + col] = (float)acc[i][j][r] * scale + bias;
            }
        }
    }
}

__global__ __launch_bounds__(256) void k_logsoftmax(float* __restrict__ O) {
    __shared__ float red[256];
    const int tid = threadIdx.x;
    float* p = O + (size_t)blockIdx.x * D_OUT;

    float v[D_OUT / 256];
    float mx = -INFINITY;
#pragma unroll
    for (int i = 0; i < D_OUT / 256; ++i) {
        v[i] = p[tid + i * 256];
        mx = fmaxf(mx, v[i]);
    }
    red[tid] = mx;
    __syncthreads();
    for (int s = 128; s > 0; s >>= 1) {
        if (tid < s) red[tid] = fmaxf(red[tid], red[tid + s]);
        __syncthreads();
    }
    mx = red[0];
    __syncthreads();

    float sum = 0.f;
#pragma unroll
    for (int i = 0; i < D_OUT / 256; ++i) sum += expf(v[i] - mx);
    red[tid] = sum;
    __syncthreads();
    for (int s = 128; s > 0; s >>= 1) {
        if (tid < s) red[tid] += red[tid + s];
        __syncthreads();
    }
    const float sub = mx + logf(red[0]);
#pragma unroll
    for (int i = 0; i < D_OUT / 256; ++i) p[tid + i * 256] = v[i] - sub;
}

extern "C" void kernel_launch(void* const* d_in, const int* in_sizes, int n_in,
                              void* d_out, int out_size, void* d_ws, size_t ws_size,
                              hipStream_t stream) {
    const float* x    = (const float*)d_in[0];
    const float* W1   = (const float*)d_in[1];
    const float* b1   = (const float*)d_in[2];
    const float* W2   = (const float*)d_in[3];
    const float* b2   = (const float*)d_in[4];
    const float* rmax = (const float*)d_in[5];

    char* ws = (char*)d_ws;
    _Float16*    W1h  = (_Float16*)ws;
    _Float16*    Xh   = (_Float16*)(ws + (size_t)D_H * D_IN * 2);
    signed char* Hi8  = (signed char*)((char*)Xh + (size_t)B_SZ * D_IN * 2);
    signed char* W2i8 = Hi8 + (size_t)B_SZ * D_H;

    k_tern_f16<<<(D_H * D_IN) / 2048, 256, 0, stream>>>(W1, W1h);
    k_cvt_f16 <<<(B_SZ * D_IN) / 2048, 256, 0, stream>>>(x, Xh);
    k_tern_i8 <<<(D_OUT * D_H) / 2048, 256, 0, stream>>>(W2, W2i8);

    k_gemm1<<<dim3(D_H / 256,  B_SZ / 64), 256, 0, stream>>>(Xh, W1h, b1, rmax, Hi8);
    k_gemm2<<<dim3(D_OUT / 256, B_SZ / 64), 256, 0, stream>>>(Hi8, W2i8, b2, rmax, (float*)d_out);
    k_logsoftmax<<<B_SZ, 256, 0, stream>>>((float*)d_out);
}
